// HET_HGTLayerHetero_7876970021040
// MI455X (gfx1250) — matrix operand order repaired and checked by reading, not compiled, not run on hardware
//
#include <hip/hip_runtime.h>
#include <hip/hip_bf16.h>
#include <stdint.h>

#define N_NODES  20000
#define N_EDGES  320000
#define IN_DIM   256
#define OUT_DIM  256
#define NH       4
#define DK       64
#define N_NTYPES 4
#define N_RELS   8
#define LDT      272   // padded LDS row stride (elements): 544B rows, 16B aligned

typedef __attribute__((ext_vector_type(16))) __bf16 v16bf;
typedef __attribute__((ext_vector_type(8)))  __bf16 v8bf;
typedef __attribute__((ext_vector_type(8)))  float  v8f;
typedef __attribute__((ext_vector_type(4)))  float  f32x4;

// ---------------- WMMA fragment helpers (CDNA5 16x16x32 bf16 layouts) ----------------
// A 16x32 from LDS: per lane two 16B contiguous runs -> 2x ds_load_b128.
static __device__ inline v16bf load_a_frag_lds(const __bf16* base, int ldm, int k0, int lane) {
  const int half = lane >> 4;
  const int m    = lane & 15;
  const __bf16* r = base + m * ldm + k0 + half * 8;
  v8bf lo = *(const v8bf*)(r);
  v8bf hi = *(const v8bf*)(r + 16);
  return __builtin_shufflevector(lo, hi, 0, 1, 2, 3, 4, 5, 6, 7,
                                 8, 9, 10, 11, 12, 13, 14, 15);
}

// B fragment from pre-swizzled bf16 weights: 32x16 K-tile stored as 32 lanes x 16
// contiguous bf16 (1KB tile). One aligned 32B load per lane (2x global_load_b128).
static __device__ inline v16bf load_b_frag_sw(const __bf16* base, int nK, int kk, int nt, int lane) {
  return *(const v16bf*)(base + (((size_t)(nt * nK + kk)) << 9) + (lane << 4));
}

static __device__ inline v8f wmma_bf16(v16bf a, v16bf b, v8f c) {
  return __builtin_amdgcn_wmma_f32_16x16x32_bf16(false, a, false, b, (short)0, c, false, false);
}

// monotonic f32 <-> u32 encoding so segment-max can use atomic_max_u32
static __device__ inline unsigned enc_f32(float f) {
  unsigned u = __float_as_uint(f);
  return (u & 0x80000000u) ? ~u : (u | 0x80000000u);
}
static __device__ inline float dec_f32(unsigned u) {
  unsigned b = (u & 0x80000000u) ? (u & 0x7FFFFFFFu) : ~u;
  return __uint_as_float(b);
}

// f32x4 pair -> packed v8bf
static __device__ inline v8bf pack_bf8(f32x4 x0, f32x4 x1) {
  v8bf t;
#pragma unroll
  for (int j = 0; j < 4; ++j) { t[j] = (__bf16)x0[j]; t[4 + j] = (__bf16)x1[j]; }
  return t;
}

// ---------------- K0: zero scratch ----------------
__global__ void zero_kernel(uint32_t* p, long n) {
  long i = (long)blockIdx.x * blockDim.x + threadIdx.x;
  if (i < n) p[i] = 0u;
}

// ---------------- K0b: pre-swizzle f32 row-major weights -> bf16 B-fragment layout ----
__global__ void swizzle_wb_kernel(const float* __restrict__ src, __bf16* __restrict__ dst,
                                  long total, int K, int N) {
  long i = (long)blockIdx.x * blockDim.x + threadIdx.x;
  if (i >= total) return;
  const int per = K * N;
  const int nK  = K >> 5;
  const long mat = i / per;
  const int  r   = (int)(i - mat * per);
  const int  j    = r & 15;
  const int  lane = (r >> 4) & 31;
  const int  tile = r >> 9;
  const int  kk   = tile % nK;
  const int  nt   = tile / nK;
  const int  k    = kk * 32 + (lane >> 4) * 8 + (j & 7) + (j >> 3) * 16;
  const int  col  = nt * 16 + (lane & 15);
  dst[i] = (__bf16)src[mat * per + (size_t)k * N + col];
}

// ---------------- K1: single-block counting sort of nodes by type + tile table ----------------
__global__ __launch_bounds__(1024)
void sort_nodes_kernel(const int* __restrict__ ntype, int* __restrict__ perm,
                       int* __restrict__ meta, int* __restrict__ tileType,
                       int* __restrict__ tileBase, int* __restrict__ tileCnt) {
  __shared__ int cnt[N_NTYPES];
  __shared__ int cur[N_NTYPES];
  const int tid = threadIdx.x;
  if (tid < N_NTYPES) cnt[tid] = 0;
  __syncthreads();
  for (int i = tid; i < N_NODES; i += 1024) atomicAdd(&cnt[ntype[i]], 1);
  __syncthreads();
  if (tid == 0) {
    int o = 0, nt = 0;
    for (int t = 0; t < N_NTYPES; ++t) {
      cur[t] = o;
      const int c = cnt[t];
      const int tiles = (c + 15) >> 4;
      for (int j = 0; j < tiles; ++j) {
        tileType[nt] = t;
        tileBase[nt] = o + j * 16;
        tileCnt[nt]  = (c - j * 16 < 16) ? (c - j * 16) : 16;
        ++nt;
      }
      o += c;
    }
    meta[0] = nt;
  }
  __syncthreads();
  for (int i = tid; i < N_NODES; i += 1024) {
    const int t = ntype[i];
    const int p = atomicAdd(&cur[t], 1);
    perm[p] = i;
  }
}

// ---------------- K2: typed K/Q/V projections (bf16 WMMA, f32 accum) ----------------
__global__ __launch_bounds__(256)
void proj_kqv_kernel(const float* __restrict__ h,
                     const __bf16* __restrict__ kWs, const __bf16* __restrict__ qWs,
                     const __bf16* __restrict__ vWs,
                     const int* __restrict__ perm, const int* __restrict__ meta,
                     const int* __restrict__ tileType, const int* __restrict__ tileBase,
                     const int* __restrict__ tileCnt,
                     float* __restrict__ Kf, __bf16* __restrict__ Qb, __bf16* __restrict__ Vb) {
  const int bid = blockIdx.x;
  if (bid >= meta[0]) return;
  const int ty = tileType[bid], base = tileBase[bid], cnt = tileCnt[bid];

  __shared__ int nodes[16];
  __shared__ __align__(16) __bf16 hT[16 * LDT];

  const int tid = threadIdx.x;
  if (tid < 16) nodes[tid] = perm[base + ((tid < cnt) ? tid : (cnt - 1))];
  __syncthreads();
  {
    const int row = tid >> 4, c0 = (tid & 15) * 16;
    const f32x4* src = (const f32x4*)(h + (size_t)nodes[row] * IN_DIM + c0);
    f32x4 x0 = src[0], x1 = src[1], x2 = src[2], x3 = src[3];
    *(v8bf*)(hT + row * LDT + c0)     = pack_bf8(x0, x1);
    *(v8bf*)(hT + row * LDT + c0 + 8) = pack_bf8(x2, x3);
  }
  __syncthreads();

  const int wave = tid >> 5, lane = tid & 31, half = lane >> 4, n = lane & 15;
#pragma unroll
  for (int jp = 0; jp < 3; ++jp) {
    const int idx0 = wave * 6 + jp * 2;         // 48 = 3 matrices * 16 col-tiles
    const int idx1 = idx0 + 1;
    const int mat0 = idx0 >> 4, n00 = (idx0 & 15) * 16;
    const int mat1 = idx1 >> 4, n01 = (idx1 & 15) * 16;
    const __bf16* W0 = (mat0 == 0 ? kWs : (mat0 == 1 ? qWs : vWs)) + (size_t)ty * IN_DIM * OUT_DIM;
    const __bf16* W1 = (mat1 == 0 ? kWs : (mat1 == 1 ? qWs : vWs)) + (size_t)ty * IN_DIM * OUT_DIM;
    v8f acc0 = {}, acc1 = {};
#pragma unroll
    for (int kk = 0; kk < 8; ++kk) {            // K = 256; A shared by both chains
      v16bf a  = load_a_frag_lds(hT, LDT, kk * 32, lane);
      v16bf b0 = load_b_frag_sw(W0, 8, kk, n00 >> 4, lane);
      v16bf b1 = load_b_frag_sw(W1, 8, kk, n01 >> 4, lane);
      acc0 = wmma_bf16(a, b0, acc0);            // two independent chains hide the
      acc1 = wmma_bf16(a, b1, acc1);            // WMMA->WMMA accumulation hazard
    }
#pragma unroll
    for (int i = 0; i < 8; ++i) {
      const int m = i + 8 * half;
      if (m < cnt) {
        const size_t o0 = (size_t)nodes[m] * OUT_DIM + n00 + n;
        const size_t o1 = (size_t)nodes[m] * OUT_DIM + n01 + n;
        if (mat0 == 0)      Kf[o0] = acc0[i];
        else if (mat0 == 1) Qb[o0] = (__bf16)acc0[i];
        else                Vb[o0] = (__bf16)acc0[i];
        if (mat1 == 0)      Kf[o1] = acc1[i];
        else if (mat1 == 1) Qb[o1] = (__bf16)acc1[i];
        else                Vb[o1] = (__bf16)acc1[i];
      }
    }
  }
}

// ---------------- K3: per-edge attention score + segment max ----------------
__global__ __launch_bounds__(256)
void edge_score_kernel(const __bf16* __restrict__ Qb, const float* __restrict__ Kf,
                       const __bf16* __restrict__ attWs, const float* __restrict__ relPri,
                       const int* __restrict__ etype, const int* __restrict__ row_idx,
                       const int* __restrict__ col_idx,
                       float* __restrict__ score, unsigned* __restrict__ menc) {
  const int e0 = blockIdx.x * 16;
  __shared__ int er[16], ec[16], et[16];
  __shared__ __align__(16) __bf16 qT[16 * LDT];
  __shared__ __align__(16) float awq[16 * LDT];

  const int tid = threadIdx.x;
  if (tid < 16) {
    int e = e0 + tid; if (e >= N_EDGES) e = N_EDGES - 1;
    er[tid] = row_idx[e]; ec[tid] = col_idx[e]; et[tid] = etype[e];
  }
  __syncthreads();
  {
    const int row = tid >> 4, c0 = (tid & 15) * 16;
    const v8bf* src = (const v8bf*)(Qb + (size_t)ec[row] * OUT_DIM + c0);
    *(v8bf*)(qT + row * LDT + c0)     = src[0];
    *(v8bf*)(qT + row * LDT + c0 + 8) = src[1];
  }
  __syncthreads();

  const int wave = tid >> 5, lane = tid & 31, half = lane >> 4, n = lane & 15;
  const int rmin = et[0], rmax = et[15];  // etype is sorted -> almost always rmin==rmax

  for (int r = rmin; r <= rmax; ++r) {
    {
      // wave's tile pair 2w,2w+1 always lies within one head -> shared A fragments
      const int n00 = wave * 32, n01 = n00 + 16;
      const int head = n00 >> 6;
      const __bf16* B = attWs + (((size_t)r * NH + head) << 12);   // 64*64 per (r,head)
      v8f acc0 = {}, acc1 = {};
#pragma unroll
      for (int kk = 0; kk < 2; ++kk) {          // K = 64
        v16bf a  = load_a_frag_lds(qT, LDT, head * DK + kk * 32, lane);
        v16bf b0 = load_b_frag_sw(B, 2, kk, (n00 & 63) >> 4, lane);
        v16bf b1 = load_b_frag_sw(B, 2, kk, (n01 & 63) >> 4, lane);
        acc0 = wmma_bf16(a, b0, acc0);
        acc1 = wmma_bf16(a, b1, acc1);
      }
#pragma unroll
      for (int i = 0; i < 8; ++i) {
        awq[(i + 8 * half) * LDT + n00 + n] = acc0[i];
        awq[(i + 8 * half) * LDT + n01 + n] = acc1[i];
      }
    }
    __syncthreads();
    {
      // 256 threads: (edge m, head hh, quarter qq) each does 16 MACs (vectorized),
      // then 4-wide butterfly reduce across adjacent lanes of the same wave32.
      const int m  = tid >> 4;
      const int hh = (tid >> 2) & 3;
      const int qq = tid & 3;
      const f32x4* ar = (const f32x4*)(awq + m * LDT + hh * DK + qq * 16);
      const f32x4* kr = (const f32x4*)(Kf + (size_t)er[m] * OUT_DIM + hh * DK + qq * 16);
      float s = 0.f;
#pragma unroll
      for (int p = 0; p < 4; ++p) {
        f32x4 av = ar[p], kv = kr[p];
#pragma unroll
        for (int d = 0; d < 4; ++d) s += av[d] * kv[d];
      }
      s += __shfl_xor(s, 1, 32);
      s += __shfl_xor(s, 2, 32);
      const int e = e0 + m;
      if (qq == 0 && e < N_EDGES && et[m] == r) {
        s *= relPri[r * NH + hh] * 0.125f;      // / sqrt(64)
        score[(size_t)e * NH + hh] = s;
        atomicMax(&menc[(size_t)ec[m] * NH + hh], enc_f32(s));
      }
    }
    __syncthreads();
  }
}

// ---------------- K4: softmax denominators ----------------
__global__ void denom_kernel(const float* __restrict__ score, const unsigned* __restrict__ menc,
                             const int* __restrict__ col_idx, float* __restrict__ denom) {
  long i = (long)blockIdx.x * blockDim.x + threadIdx.x;
  if (i >= (long)N_EDGES * NH) return;
  const int e = (int)(i >> 2), hh = (int)(i & 3);
  const int c = col_idx[e];
  const float m = dec_f32(menc[(size_t)c * NH + hh]);
  atomicAdd(&denom[(size_t)c * NH + hh], __expf(score[i] - m));
}

// ---------------- K5: recompute msg via WMMA, scatter alpha*msg into agg ----------------
__global__ __launch_bounds__(256)
void edge_agg_kernel(const __bf16* __restrict__ Vb, const __bf16* __restrict__ msgWs,
                     const float* __restrict__ score, const unsigned* __restrict__ menc,
                     const float* __restrict__ denom,
                     const int* __restrict__ etype, const int* __restrict__ row_idx,
                     const int* __restrict__ col_idx, float* __restrict__ agg) {
  const int e0 = blockIdx.x * 16;
  __shared__ int er[16], ec[16], et[16];
  __shared__ float alphaS[16 * 4];
  __shared__ __align__(16) __bf16 vT[16 * LDT];

  const int tid = threadIdx.x;
  if (tid < 16) {
    int e = e0 + tid; if (e >= N_EDGES) e = N_EDGES - 1;
    er[tid] = row_idx[e]; ec[tid] = col_idx[e]; et[tid] = etype[e];
  }
  __syncthreads();
  {
    const int row = tid >> 4, c0 = (tid & 15) * 16;
    const v8bf* src = (const v8bf*)(Vb + (size_t)er[row] * OUT_DIM + c0);
    *(v8bf*)(vT + row * LDT + c0)     = src[0];
    *(v8bf*)(vT + row * LDT + c0 + 8) = src[1];
  }
  if (tid < 64) {
    const int m = tid >> 2, hh = tid & 3;
    const int e = e0 + m;
    float al = 0.f;
    if (e < N_EDGES) {
      const int c = ec[m];
      const float mval = dec_f32(menc[(size_t)c * NH + hh]);
      const float ex = __expf(score[(size_t)e * NH + hh] - mval);
      al = ex / denom[(size_t)c * NH + hh];
    }
    alphaS[m * 4 + hh] = al;
  }
  __syncthreads();

  const int wave = tid >> 5, lane = tid & 31, half = lane >> 4, n = lane & 15;
  const int rmin = et[0], rmax = et[15];
  for (int r = rmin; r <= rmax; ++r) {
    const int n00 = wave * 32, n01 = n00 + 16;
    const int head = n00 >> 6;
    const __bf16* B = msgWs + (((size_t)r * NH + head) << 12);
    v8f acc0 = {}, acc1 = {};
#pragma unroll
    for (int kk = 0; kk < 2; ++kk) {
      v16bf a  = load_a_frag_lds(vT, LDT, head * DK + kk * 32, lane);
      v16bf b0 = load_b_frag_sw(B, 2, kk, (n00 & 63) >> 4, lane);
      v16bf b1 = load_b_frag_sw(B, 2, kk, (n01 & 63) >> 4, lane);
      acc0 = wmma_bf16(a, b0, acc0);
      acc1 = wmma_bf16(a, b1, acc1);
    }
#pragma unroll
    for (int i = 0; i < 8; ++i) {
      const int m = i + 8 * half;
      const int e = e0 + m;
      if (e < N_EDGES && et[m] == r) {
        const float al = alphaS[m * 4 + head];
        atomicAdd(&agg[(size_t)ec[m] * OUT_DIM + n00 + n], al * acc0[i]);
        atomicAdd(&agg[(size_t)ec[m] * OUT_DIM + n01 + n], al * acc1[i]);
      }
    }
  }
}

// ---------------- K6: typed output projection + sigmoid(skip) scale ----------------
__global__ __launch_bounds__(256)
void out_proj_kernel(const float* __restrict__ agg, const __bf16* __restrict__ aWs,
                     const float* __restrict__ skip,
                     const int* __restrict__ perm, const int* __restrict__ meta,
                     const int* __restrict__ tileType, const int* __restrict__ tileBase,
                     const int* __restrict__ tileCnt, float* __restrict__ out) {
  const int bid = blockIdx.x;
  if (bid >= meta[0]) return;
  const int ty = tileType[bid], base = tileBase[bid], cnt = tileCnt[bid];

  __shared__ int nodes[16];
  __shared__ __align__(16) __bf16 gT[16 * LDT];

  const int tid = threadIdx.x;
  if (tid < 16) nodes[tid] = perm[base + ((tid < cnt) ? tid : (cnt - 1))];
  __syncthreads();
  {
    const int row = tid >> 4, c0 = (tid & 15) * 16;
    const f32x4* src = (const f32x4*)(agg + (size_t)nodes[row] * OUT_DIM + c0);
    f32x4 x0 = src[0], x1 = src[1], x2 = src[2], x3 = src[3];
    *(v8bf*)(gT + row * LDT + c0)     = pack_bf8(x0, x1);
    *(v8bf*)(gT + row * LDT + c0 + 8) = pack_bf8(x2, x3);
  }
  __syncthreads();

  const float sg = 1.f / (1.f + __expf(-skip[ty]));
  const int wave = tid >> 5, lane = tid & 31, half = lane >> 4, n = lane & 15;
  const __bf16* W = aWs + (size_t)ty * OUT_DIM * OUT_DIM;
  const int n00 = wave * 32, n01 = n00 + 16;
  v8f acc0 = {}, acc1 = {};
#pragma unroll
  for (int kk = 0; kk < 8; ++kk) {
    v16bf a  = load_a_frag_lds(gT, LDT, kk * 32, lane);
    v16bf b0 = load_b_frag_sw(W, 8, kk, n00 >> 4, lane);
    v16bf b1 = load_b_frag_sw(W, 8, kk, n01 >> 4, lane);
    acc0 = wmma_bf16(a, b0, acc0);
    acc1 = wmma_bf16(a, b1, acc1);
  }
#pragma unroll
  for (int i = 0; i < 8; ++i) {
    const int m = i + 8 * half;
    if (m < cnt) {
      out[(size_t)nodes[m] * OUT_DIM + n00 + n] = sg * acc0[i];
      out[(size_t)nodes[m] * OUT_DIM + n01 + n] = sg * acc1[i];
    }
  }
}

// ---------------- launcher ----------------
extern "C" void kernel_launch(void* const* d_in, const int* in_sizes, int n_in,
                              void* d_out, int out_size, void* d_ws, size_t ws_size,
                              hipStream_t stream) {
  const float* h       = (const float*)d_in[0];
  const float* kW      = (const float*)d_in[1];
  const float* qW      = (const float*)d_in[2];
  const float* vW      = (const float*)d_in[3];
  const float* aW      = (const float*)d_in[4];
  const float* relAtt  = (const float*)d_in[5];
  const float* relMsg  = (const float*)d_in[6];
  const float* relPri  = (const float*)d_in[7];
  const float* skip    = (const float*)d_in[8];
  const int*   ntype   = (const int*)d_in[9];
  const int*   etype   = (const int*)d_in[10];
  const int*   row_idx = (const int*)d_in[11];
  const int*   col_idx = (const int*)d_in[12];
  float* out = (float*)d_out;

  // carve workspace
  char* ws = (char*)d_ws;
  size_t off = 0;
  auto alloc = [&](size_t bytes) -> void* {
    off = (off + 255) & ~(size_t)255;
    void* p = ws + off;
    off += bytes;
    return p;
  };
  const int MAXT = N_NODES / 16 + N_NTYPES;
  const long projW = (long)N_NTYPES * IN_DIM * OUT_DIM;     // 262144 elems per tensor
  const long relW  = (long)N_RELS * NH * DK * DK;           // 131072 elems per tensor

  int*      meta     = (int*)alloc(16 * sizeof(int));
  int*      perm     = (int*)alloc((size_t)N_NODES * sizeof(int));
  int*      tileType = (int*)alloc((size_t)MAXT * sizeof(int));
  int*      tileBase = (int*)alloc((size_t)MAXT * sizeof(int));
  int*      tileCnt  = (int*)alloc((size_t)MAXT * sizeof(int));
  __bf16*   kWs      = (__bf16*)alloc((size_t)projW * sizeof(__bf16));
  __bf16*   qWs      = (__bf16*)alloc((size_t)projW * sizeof(__bf16));
  __bf16*   vWs      = (__bf16*)alloc((size_t)projW * sizeof(__bf16));
  __bf16*   aWs      = (__bf16*)alloc((size_t)projW * sizeof(__bf16));
  __bf16*   attWs    = (__bf16*)alloc((size_t)relW * sizeof(__bf16));
  __bf16*   msgWs    = (__bf16*)alloc((size_t)relW * sizeof(__bf16));
  float*    Kf       = (float*)alloc((size_t)N_NODES * OUT_DIM * sizeof(float));
  __bf16*   Qb       = (__bf16*)alloc((size_t)N_NODES * OUT_DIM * sizeof(__bf16));
  __bf16*   Vb       = (__bf16*)alloc((size_t)N_NODES * OUT_DIM * sizeof(__bf16));
  float*    scoreB   = (float*)alloc((size_t)N_EDGES * NH * sizeof(float));
  // contiguous zero region: menc | denom | agg
  const long nMenc  = (long)N_NODES * NH;
  const long nDen   = (long)N_NODES * NH;
  const long nAgg   = (long)N_NODES * OUT_DIM;
  const long nZero  = nMenc + nDen + nAgg;
  uint32_t* zeroP   = (uint32_t*)alloc((size_t)nZero * 4);
  unsigned* menc    = (unsigned*)zeroP;
  float*    denom   = (float*)(zeroP + nMenc);
  float*    agg     = (float*)(zeroP + nMenc + nDen);

  const int nETiles = (N_EDGES + 15) / 16;

  zero_kernel<<<(int)((nZero + 255) / 256), 256, 0, stream>>>(zeroP, nZero);
  // pre-swizzle all weights into bf16 WMMA B-fragment layout (cheap, ~1.3M elems)
  swizzle_wb_kernel<<<(int)((projW + 255) / 256), 256, 0, stream>>>(kW, kWs, projW, IN_DIM, OUT_DIM);
  swizzle_wb_kernel<<<(int)((projW + 255) / 256), 256, 0, stream>>>(qW, qWs, projW, IN_DIM, OUT_DIM);
  swizzle_wb_kernel<<<(int)((projW + 255) / 256), 256, 0, stream>>>(vW, vWs, projW, IN_DIM, OUT_DIM);
  swizzle_wb_kernel<<<(int)((projW + 255) / 256), 256, 0, stream>>>(aW, aWs, projW, OUT_DIM, OUT_DIM);
  swizzle_wb_kernel<<<(int)((relW + 255) / 256), 256, 0, stream>>>(relAtt, attWs, relW, DK, DK);
  swizzle_wb_kernel<<<(int)((relW + 255) / 256), 256, 0, stream>>>(relMsg, msgWs, relW, DK, DK);

  sort_nodes_kernel<<<1, 1024, 0, stream>>>(ntype, perm, meta, tileType, tileBase, tileCnt);
  proj_kqv_kernel<<<MAXT, 256, 0, stream>>>(h, kWs, qWs, vWs, perm, meta, tileType, tileBase,
                                            tileCnt, Kf, Qb, Vb);
  edge_score_kernel<<<nETiles, 256, 0, stream>>>(Qb, Kf, attWs, relPri, etype, row_idx,
                                                 col_idx, scoreB, menc);
  denom_kernel<<<(int)(((long)N_EDGES * NH + 255) / 256), 256, 0, stream>>>(scoreB, menc,
                                                                            col_idx, denom);
  edge_agg_kernel<<<nETiles, 256, 0, stream>>>(Vb, msgWs, scoreB, menc, denom, etype,
                                               row_idx, col_idx, agg);
  out_proj_kernel<<<MAXT, 256, 0, stream>>>(agg, aWs, skip, perm, meta, tileType, tileBase,
                                            tileCnt, out);
}
